// InstructionEncoder_46024869543965
// MI455X (gfx1250) — matrix operand order repaired and checked by reading, not compiled, not run on hardware
//
#include <hip/hip_runtime.h>
#include <hip/hip_bf16.h>
#include <math.h>
#include <stdint.h>

// Problem constants
#define Bsz   128
#define Tsz   256
#define EMB   300
#define KX0   320          // EMBED padded to multiple of 32
#define Hn    512
#define NZ    2048         // 4*H gate columns

typedef __attribute__((ext_vector_type(16))) __bf16 v16bf;
typedef __attribute__((ext_vector_type(8)))  float  v8f;

union FragBF { v16bf v; uint4 q[2]; };

// ---- CDNA5 async copy global->LDS (ASYNCcnt) -------------------------------
// GV form: LDS address from the vdst VGPR, global address from a 64-bit VGPR pair.
__device__ __forceinline__ void async_copy_b128(uint32_t lds_off, const void* gaddr) {
    asm volatile("global_load_async_to_lds_b128 %0, %1, off"
                 :: "v"(lds_off), "v"((unsigned long long)(uintptr_t)gaddr)
                 : "memory");
}
__device__ __forceinline__ void wait_async0() {
    asm volatile("s_wait_asynccnt 0x0" ::: "memory");
}

// ---------------------------------------------------------------- zero state
__global__ void zero_u4_kernel(uint4* __restrict__ p, int n) {
    int i = blockIdx.x * blockDim.x + threadIdx.x;
    if (i < n) p[i] = make_uint4(0u, 0u, 0u, 0u);
}

// ------------------------------------------------------- embedding -> bf16 x0
// x0 layout: [T, B, KX0] bf16, zero-padded beyond EMB.
__global__ void embed_gather_kernel(const int* __restrict__ tok,
                                    const float* __restrict__ emb,
                                    __bf16* __restrict__ x0, int total) {
    int i = blockIdx.x * blockDim.x + threadIdx.x;
    if (i >= total) return;
    int k  = i % KX0;
    int tb = i / KX0;          // t*B + b
    int b  = tb % Bsz;
    int t  = tb / Bsz;
    float v = 0.f;
    if (k < EMB) {
        int id = tok[b * Tsz + t];
        v = emb[(size_t)id * EMB + k];
    }
    x0[i] = (__bf16)v;
}

// ------------------------------------------------ pack [Wk;Wr] -> B fragments
// Packed element order: ((nt*KT + kt)*32 + lane)*16 + e, e = 2*vgpr + pair.
// B 32x16 bf16 fragment: lane<16 -> n = nt*16+lane, K = kt*32 + e (0..15)
//                        lane>=16 -> n = nt*16+lane-16, K = kt*32 + 16 + e
__global__ void pack_w_kernel(const float* __restrict__ Wk,
                              const float* __restrict__ Wr,
                              __bf16* __restrict__ dst,
                              int KXp, int KXreal, int KT, int total) {
    int i = blockIdx.x * blockDim.x + threadIdx.x;
    if (i >= total) return;
    int e    = i & 15;
    int lane = (i >> 4) & 31;
    int tile = i >> 9;
    int kt   = tile % KT;
    int nt   = tile / KT;
    int n    = nt * 16 + (lane & 15);
    int k    = kt * 32 + ((lane >> 4) << 4) + e;
    float v;
    if (k < KXp) v = (k < KXreal) ? Wk[(size_t)k * NZ + n] : 0.f;
    else         v = Wr[(size_t)(k - KXp) * NZ + n];
    dst[i] = (__bf16)v;
}

// ---------------------------------------------------------------- LSTM step
struct DirArgs {
    const __bf16* x;     // [T,B,KX] layer input sequence (bf16)
    const __bf16* wpk;   // packed [Wk;Wr] fragments
    const float*  bias;  // [2048]
    const __bf16* hin;   // [B,H] carried h (read)
    __bf16*       hout;  // [B,H] carried h (write)
    float*        cst;   // [B,H] cell state f32
    __bf16*       ybf;   // [T,B,H] sequence output bf16 (layer 0)
    int           t;     // time index for this direction
};

template <int KX, bool LAST>
__global__ __launch_bounds__(256) void lstm_step_kernel(
        DirArgs a0, DirArgs a1, const int* __restrict__ tok,
        float* __restrict__ out_seq,   // [B,T,2H] (layer 1 only)
        float* __restrict__ st_h,      // [B,2H] slice of states_h for this layer
        float* __restrict__ st_c) {    // [B,2H] slice of states_c
    constexpr int KT = (KX + Hn) / 32;
    __shared__ unsigned char shb[2 * 8192];   // double-buffered 8 B tiles (8 KB each)

    const int dir  = blockIdx.x >> 4;        // 0 = fwd, 1 = bwd
    const int cg   = blockIdx.x & 15;        // 32 h-columns per block
    const DirArgs A = dir ? a1 : a0;
    const int t    = A.t;
    const int tid  = threadIdx.x;
    const int lane = tid & 31;
    const int w    = tid >> 5;               // 8 waves
    const int m0   = w * 16;                 // 16-row batch strip per wave
    const int n0   = cg * 32;

    // 8 accumulator tiles: [i0,i1,f0,f1,g0,g1,o0,o1]
    v8f acc[8];
    #pragma unroll
    for (int j = 0; j < 8; ++j)
        #pragma unroll
        for (int r = 0; r < 8; ++r) acc[j][r] = 0.f;

    int nt[8];
    #pragma unroll
    for (int j = 0; j < 8; ++j) nt[j] = (j >> 1) * 32 + cg * 2 + (j & 1);

    // Async-copy chunk assignment: 8 KB/iter = 512 x 16B, 2 chunks per thread.
    // chunk c: tile j = c>>6, byte offset within tile = (c&63)*16.
    const int c0j = tid >> 6,          c0o = (tid & 63) * 16;
    const int c1j = (tid + 256) >> 6,  c1o = ((tid + 256) & 63) * 16;
    const size_t src0base = ((size_t)nt[c0j] * KT) * 1024 + c0o;
    const size_t src1base = ((size_t)nt[c1j] * KT) * 1024 + c1o;
    const char* wbytes = (const char*)A.wpk;

    // prefetch kt = 0 into buffer 0
    {
        uint32_t d0 = (uint32_t)(uintptr_t)(shb + c0j * 1024 + c0o);
        uint32_t d1 = (uint32_t)(uintptr_t)(shb + c1j * 1024 + c1o);
        async_copy_b128(d0, wbytes + src0base);
        async_copy_b128(d1, wbytes + src1base);
        wait_async0();
    }
    __syncthreads();

    const int mrow = m0 + (lane & 15);
    const int ksub = (lane >> 4) << 4;       // 0 or 16 (A-fragment K split)

    for (int kt = 0; kt < KT; ++kt) {
        // kick off async copy of next K step's B tiles into the other buffer
        if (kt + 1 < KT) {
            const int par = (kt + 1) & 1;
            uint32_t d0 = (uint32_t)(uintptr_t)(shb + par * 8192 + c0j * 1024 + c0o);
            uint32_t d1 = (uint32_t)(uintptr_t)(shb + par * 8192 + c1j * 1024 + c1o);
            async_copy_b128(d0, wbytes + src0base + (size_t)(kt + 1) * 1024);
            async_copy_b128(d1, wbytes + src1base + (size_t)(kt + 1) * 1024);
        }

        const int k0 = kt * 32;
        const __bf16* ap;
        if (k0 < KX) ap = A.x  + ((size_t)t * Bsz + mrow) * KX + (k0 + ksub);
        else         ap = A.hin + (size_t)mrow * Hn + (k0 - KX + ksub);
        FragBF af;
        af.q[0] = *(const uint4*)(ap);        // K ksub..ksub+7
        af.q[1] = *(const uint4*)(ap + 8);    // K ksub+8..ksub+15

        const unsigned char* bbase = shb + (kt & 1) * 8192 + lane * 32;
        #pragma unroll
        for (int j = 0; j < 8; ++j) {
            const uint4* bq = (const uint4*)(bbase + j * 1024);
            FragBF bf;
            bf.q[0] = bq[0];
            bf.q[1] = bq[1];
            acc[j] = __builtin_amdgcn_wmma_f32_16x16x32_bf16(
                false, af.v, false, bf.v, (short)0, acc[j], false, false);
        }

        wait_async0();       // our prefetch for kt+1 has landed in LDS
        __syncthreads();     // everyone done reading buf[kt&1] / writing buf[(kt+1)&1]
    }

    // Gates + state update. C/D layout: VGPR r -> M = r + 8*(lane>=16), N = lane%16.
    const int nl  = lane & 15;
    const int rhi = (lane >> 4) << 3;
    #pragma unroll
    for (int jj = 0; jj < 2; ++jj) {
        const int hc = n0 + jj * 16 + nl;
        const float bi = A.bias[0 * Hn + hc];
        const float bff = A.bias[1 * Hn + hc];
        const float bg = A.bias[2 * Hn + hc];
        const float bo = A.bias[3 * Hn + hc];
        #pragma unroll
        for (int r = 0; r < 8; ++r) {
            const int m = m0 + r + rhi;
            const float iv = acc[0 + jj][r] + bi;
            const float fv = acc[2 + jj][r] + bff;
            const float gv = acc[4 + jj][r] + bg;
            const float ov = acc[6 + jj][r] + bo;
            const float co = A.cst[(size_t)m * Hn + hc];
            const float si = 1.f / (1.f + __expf(-iv));
            const float sf = 1.f / (1.f + __expf(-fv));
            const float so = 1.f / (1.f + __expf(-ov));
            const float cn = sf * co + si * tanhf(gv);
            const float hnew = so * tanhf(cn);
            const bool valid = (tok[m * Tsz + t] != 0);
            const float hk = valid ? hnew : (float)A.hin[(size_t)m * Hn + hc];
            const float ck = valid ? cn : co;
            A.cst[(size_t)m * Hn + hc]  = ck;
            A.hout[(size_t)m * Hn + hc] = (__bf16)hk;
            if (LAST) {
                out_seq[((size_t)m * Tsz + t) * 1024 + (size_t)dir * Hn + hc] = hk;
            } else {
                A.ybf[((size_t)t * Bsz + m) * Hn + hc] = (__bf16)hk;
            }
            st_h[(size_t)m * 1024 + (size_t)dir * Hn + hc] = hk;
            st_c[(size_t)m * 1024 + (size_t)dir * Hn + hc] = ck;
        }
    }
}

// ---------------------------------------------------------------- launcher
extern "C" void kernel_launch(void* const* d_in, const int* in_sizes, int n_in,
                              void* d_out, int out_size, void* d_ws, size_t ws_size,
                              hipStream_t stream) {
    (void)in_sizes; (void)n_in; (void)out_size; (void)ws_size;
    const int*   tok   = (const int*)d_in[0];
    const float* emb   = (const float*)d_in[1];
    const float* Wk_f0 = (const float*)d_in[2];
    const float* Wr_f0 = (const float*)d_in[3];
    const float* b_f0  = (const float*)d_in[4];
    const float* Wk_f1 = (const float*)d_in[5];
    const float* Wr_f1 = (const float*)d_in[6];
    const float* b_f1  = (const float*)d_in[7];
    const float* Wk_b0 = (const float*)d_in[8];
    const float* Wr_b0 = (const float*)d_in[9];
    const float* b_b0  = (const float*)d_in[10];
    const float* Wk_b1 = (const float*)d_in[11];
    const float* Wr_b1 = (const float*)d_in[12];
    const float* b_b1  = (const float*)d_in[13];

    char* ws = (char*)d_ws;
    size_t off = 0;
    auto take = [&](size_t bytes) -> char* {
        char* p = ws + off;
        off = (off + bytes + 255) & ~(size_t)255;
        return p;
    };

    __bf16* x0  = (__bf16*)take((size_t)Tsz * Bsz * KX0 * 2);
    __bf16* y0f = (__bf16*)take((size_t)Tsz * Bsz * Hn * 2);
    __bf16* y0b = (__bf16*)take((size_t)Tsz * Bsz * Hn * 2);

    const int KT0 = (KX0 + Hn) / 32;   // 26
    const int KT1 = (Hn  + Hn) / 32;   // 32
    const int packTot0 = (KX0 + Hn) * NZ;   // 832*2048
    const int packTot1 = (Hn  + Hn) * NZ;   // 1024*2048
    __bf16* wp_f0 = (__bf16*)take((size_t)packTot0 * 2);
    __bf16* wp_b0 = (__bf16*)take((size_t)packTot0 * 2);
    __bf16* wp_f1 = (__bf16*)take((size_t)packTot1 * 2);
    __bf16* wp_b1 = (__bf16*)take((size_t)packTot1 * 2);

    // carried state zone (zeroed every call): h double-buffers (bf16) + c (f32)
    size_t state_begin = off;
    __bf16* hbuf[4][2];
    for (int dl = 0; dl < 4; ++dl)
        for (int par = 0; par < 2; ++par)
            hbuf[dl][par] = (__bf16*)take((size_t)Bsz * Hn * 2);
    float* cbuf[4];
    for (int dl = 0; dl < 4; ++dl)
        cbuf[dl] = (float*)take((size_t)Bsz * Hn * 4);
    size_t state_bytes = off - state_begin;     // 2 MB, multiple of 16

    const int zn = (int)(state_bytes / 16);
    zero_u4_kernel<<<(zn + 255) / 256, 256, 0, stream>>>((uint4*)(ws + state_begin), zn);

    const int embTot = Tsz * Bsz * KX0;
    embed_gather_kernel<<<(embTot + 255) / 256, 256, 0, stream>>>(tok, emb, x0, embTot);

    pack_w_kernel<<<(packTot0 + 255) / 256, 256, 0, stream>>>(Wk_f0, Wr_f0, wp_f0, KX0, EMB, KT0, packTot0);
    pack_w_kernel<<<(packTot0 + 255) / 256, 256, 0, stream>>>(Wk_b0, Wr_b0, wp_b0, KX0, EMB, KT0, packTot0);
    pack_w_kernel<<<(packTot1 + 255) / 256, 256, 0, stream>>>(Wk_f1, Wr_f1, wp_f1, Hn, Hn, KT1, packTot1);
    pack_w_kernel<<<(packTot1 + 255) / 256, 256, 0, stream>>>(Wk_b1, Wr_b1, wp_b1, Hn, Hn, KT1, packTot1);

    float* outseq = (float*)d_out;                                  // [B,T,1024]
    float* st_h0  = outseq + (size_t)Bsz * Tsz * 1024;              // states_h layer 0
    float* st_h1  = st_h0 + (size_t)Bsz * 1024;                     // states_h layer 1
    float* st_c0  = st_h0 + (size_t)2 * Bsz * 1024;                 // states_c layer 0
    float* st_c1  = st_c0 + (size_t)Bsz * 1024;                     // states_c layer 1

    // Layer 0: fwd (t = s) and bwd (t = T-1-s) fused per launch.
    for (int s = 0; s < Tsz; ++s) {
        DirArgs a0{ x0, wp_f0, b_f0, hbuf[0][s & 1], hbuf[0][(s + 1) & 1], cbuf[0], y0f, s };
        DirArgs a1{ x0, wp_b0, b_b0, hbuf[1][s & 1], hbuf[1][(s + 1) & 1], cbuf[1], y0b, Tsz - 1 - s };
        lstm_step_kernel<KX0, false><<<32, 256, 0, stream>>>(a0, a1, tok, nullptr, st_h0, st_c0);
    }
    // Layer 1: consumes layer-0 sequences; writes final output + states.
    for (int s = 0; s < Tsz; ++s) {
        DirArgs a0{ y0f, wp_f1, b_f1, hbuf[2][s & 1], hbuf[2][(s + 1) & 1], cbuf[2], nullptr, s };
        DirArgs a1{ y0b, wp_b1, b_b1, hbuf[3][s & 1], hbuf[3][(s + 1) & 1], cbuf[3], nullptr, Tsz - 1 - s };
        lstm_step_kernel<Hn, true><<<32, 256, 0, stream>>>(a0, a1, tok, outseq, st_h1, st_c1);
    }
}
